// Model_20074677141662
// MI455X (gfx1250) — compile-verified, matrix operand order checked by reading, not run on hardware
//
#include <hip/hip_runtime.h>
#include <hip/hip_bf16.h>

// MI455X / gfx1250, wave32. WMMA f16 16x16x32 with f32 accumulation.
// Conv-as-Toeplitz-GEMM (128x4096x2048 per channel) + projection GEMM
// (8192x2048x1440); kernel table staged to LDS via gfx1250 async-to-LDS
// (GLOBAL_LOAD_ASYNC_TO_LDS_B128, tracked by ASYNCcnt).
//
// Workspace layout (bytes):
//   mean  : 8192 f32            @ 0
//   rstd  : 8192 f32            @ 32768
//   Sg    : 64ch x 8192 f16     @ 65536        (padded Toeplitz kernel table)
//   pw16  : 1440 x 2048 f16     @ 1114112
//   A1    : 64 x 128 x 4096 f16 @ 7012352      ([un | W] packed per channel)
//   y2    : 8192 x 2048 f16     @ 74121216
//   Z     : 8192 x 1440 f32     @ 107675648
// total ~148 MB

typedef _Float16 f16;
typedef __attribute__((ext_vector_type(16))) _Float16 v16h;
typedef __attribute__((ext_vector_type(8)))  _Float16 v8h;
typedef __attribute__((ext_vector_type(8)))  float    v8f;
typedef __attribute__((ext_vector_type(4)))  unsigned int u32x4;
typedef __attribute__((ext_vector_type(4)))  int i32x4;
// AS-qualified pointee types for the gfx1250 async-to-LDS builtin
// (signature: void(int4 AS1*, int4 AS3*, imm offset, imm cpol)).
typedef __attribute__((address_space(1))) i32x4 gbl_i32x4;
typedef __attribute__((address_space(3))) i32x4 lds_i32x4;

union Frag16 {
    v16h v;
    v8h  q[2];
    _Float16 h[16];
};

#define OFF_MEAN  ((size_t)0)
#define OFF_RSTD  ((size_t)32768)
#define OFF_SG    ((size_t)65536)
#define OFF_PW    ((size_t)1114112)
#define OFF_A1    ((size_t)7012352)
#define OFF_Y2    ((size_t)74121216)
#define OFF_Z     ((size_t)107675648)

#define NB   128
#define NL   2048
#define NC   64
#define NP   720

// ---------------- zero fill (Sg padding regions must be 0) ----------------
__global__ void zero_kernel(unsigned int* p, unsigned long long n32) {
    unsigned long long i = (unsigned long long)blockIdx.x * 256ull + threadIdx.x;
    if (i < n32) p[i] = 0u;
}

// ---------------- per-(b,c) mean / rstd over time ----------------
__global__ __launch_bounds__(256) void stats_kernel(const float* __restrict__ u,
                                                    float* __restrict__ mean,
                                                    float* __restrict__ rstd) {
    __shared__ float s0[256], s1[256];
    const int b  = blockIdx.x;
    const int cc = threadIdx.x & 63;
    const int rg = threadIdx.x >> 6;          // 4 row groups
    const float* ub = u + (size_t)b * NL * NC;
    float sum = 0.f, sq = 0.f;
    for (int l = rg; l < NL; l += 4) {
        float v = ub[(size_t)l * NC + cc];
        sum += v; sq += v * v;
    }
    s0[threadIdx.x] = sum; s1[threadIdx.x] = sq;
    __syncthreads();
    if (threadIdx.x < 64) {
        float S = 0.f, Q = 0.f;
        #pragma unroll
        for (int r = 0; r < 4; ++r) { S += s0[r * 64 + cc]; Q += s1[r * 64 + cc]; }
        float m   = S / (float)NL;
        float var = (Q - (float)NL * m * m) / (float)(NL - 1) + 1e-5f;
        mean[b * NC + cc] = m;
        rstd[b * NC + cc] = rsqrtf(var);
    }
}

// ---------------- SSM kernels K, K2 -> padded per-channel table S ----------------
// S[c][m]        = K2[m][c]  for 1<=m<2048, else 0   (region 0)
// S[c][4096 + m] = K [m][c]  for 1<=m<2048, else 0   (region 1)
__global__ __launch_bounds__(64) void ssm_kernel(const float* __restrict__ C_param,
                                                 const float* __restrict__ log_dt,
                                                 const float* __restrict__ log_A_real,
                                                 const float* __restrict__ A_imag,
                                                 f16* __restrict__ Sg) {
    __shared__ float lar[64], aim[64], cre[64], cim[64];
    const int hh = blockIdx.x;      // kernel time tap 0..2047
    const int tc = threadIdx.x;     // channel 0..63 ( == 'l' in reference)
    lar[tc] = log_A_real[hh * 64 + tc];
    aim[tc] = A_imag[hh * 64 + tc];
    cre[tc] = C_param[(hh * 64 + tc) * 2 + 0];
    cim[tc] = C_param[(hh * 64 + tc) * 2 + 1];
    __syncthreads();
    const float dt = expf(log_dt[hh]);
    const float lc = (float)tc;
    float sumRe = 0.f;
    for (int n = 0; n < 64; ++n) {
        float Ar = -expf(lar[n]);
        float Ai = aim[n];
        float dr = Ar * dt, di = Ai * dt;
        // (exp(dtA) - 1) / A
        float er    = expf(dr);
        float num_r = er * cosf(di) - 1.f;
        float num_i = er * sinf(di);
        float den   = Ar * Ar + Ai * Ai;
        float q_r   = (num_r * Ar + num_i * Ai) / den;
        float q_i   = (num_i * Ar - num_r * Ai) / den;
        // Cc = C * q
        float ccr = cre[n] * q_r - cim[n] * q_i;
        float cci = cre[n] * q_i + cim[n] * q_r;
        // exp(dtA * l)
        float el  = expf(dr * lc);
        float ang = di * lc;
        float Er  = el * cosf(ang), Ei = el * sinf(ang);
        sumRe += ccr * Er - cci * Ei;
    }
    float Kv   = 2.f * sumRe;
    float ReBk = -expf(lar[0]) * lc * dt;     // Re(A[h,0]) * l * dt
    float K2v  = 2.f * Kv * ReBk;
    if (hh >= 1) {
        Sg[(size_t)tc * 8192 + hh]        = (f16)K2v;
        Sg[(size_t)tc * 8192 + 4096 + hh] = (f16)Kv;
    }
}

// ---------------- pack A1 = f16 [un | W] in (c, b, j) layout ----------------
__global__ void pack_kernel(const float* __restrict__ u, const float* __restrict__ Wn,
                            const float* __restrict__ logvar,
                            const float* __restrict__ mean, const float* __restrict__ rstd,
                            f16* __restrict__ A1) {
    size_t tid = (size_t)blockIdx.x * 256 + threadIdx.x;
    if (tid >= (size_t)NB * NL * NC) return;
    int cc = (int)(tid % NC);
    int l  = (int)((tid / NC) % NL);
    int b  = (int)(tid / ((size_t)NC * NL));
    float m  = mean[b * NC + cc];
    float rs = rstd[b * NC + cc];
    float sn = sqrtf(expf(logvar[0]));
    float un = (u[tid] - m) * rs;
    float w  = sn * Wn[tid];
    size_t base = ((size_t)cc * NB + b) * 4096;
    A1[base + l]        = (f16)un;
    A1[base + 2048 + l] = (f16)w;
}

// ---------------- proj_w -> f16 ----------------
__global__ void pwpack_kernel(const float* __restrict__ pw, f16* __restrict__ pw16) {
    size_t i = (size_t)blockIdx.x * 256 + threadIdx.x;
    if (i < (size_t)2 * NP * NL) pw16[i] = (f16)pw[i];
}

// ---------------- GEMM1: per-channel 128 x 4096 x 2048 Toeplitz GEMM ----------------
// Y[b,t,c] = sum_j A1[c,b,j] * S_c[ (j<2048 ? 2048 : 8192) + t - j ]  + un[b,t,c]*D[t]
__global__ __launch_bounds__(256) void gemm1_kernel(const f16* __restrict__ A1,
                                                    const f16* __restrict__ Sg,
                                                    const float* __restrict__ Dg,
                                                    f16* __restrict__ y2) {
    __shared__ f16 sS[8192];
    const int c  = blockIdx.x;        // channel
    const int tt = blockIdx.y;        // 128-wide output-time supertile

    // Stage the 16 KB padded kernel table into LDS.
#if __has_builtin(__builtin_amdgcn_global_load_async_to_lds_b128)
    {
        // gfx1250 async global->LDS path (ASYNCcnt): 256 threads x 4 x 16B.
        const f16* src = Sg + (size_t)c * 8192;
        #pragma unroll
        for (int pass = 0; pass < 4; ++pass) {
            const int i = pass * 256 + threadIdx.x;   // 16B chunk index, 0..1023
            __builtin_amdgcn_global_load_async_to_lds_b128(
                (gbl_i32x4*)(src + i * 8),
                (lds_i32x4*)(sS + i * 8),
                0, 0);
        }
#if __has_builtin(__builtin_amdgcn_s_wait_asynccnt)
        __builtin_amdgcn_s_wait_asynccnt(0);
#else
        asm volatile("s_wait_asynccnt 0x0" ::: "memory");
#endif
    }
#else
    {   // fallback: cooperative b128 copy through VGPRs
        const u32x4* src = (const u32x4*)(Sg + (size_t)c * 8192);
        u32x4* dst = (u32x4*)sS;
        for (int i = threadIdx.x; i < 1024; i += 256) dst[i] = src[i];
    }
#endif
    __syncthreads();

    const int wave = threadIdx.x >> 5;
    const int lane = threadIdx.x & 31;
    const int nrow = lane & 15;       // B column n / A row m within tile
    const int grp  = lane >> 4;       // lane group (K sub-range select)
    const int t0   = tt * 128 + wave * 16;

    v8f acc[8];
    const v8f vz = {0.f,0.f,0.f,0.f,0.f,0.f,0.f,0.f};
    #pragma unroll
    for (int i = 0; i < 8; ++i) acc[i] = vz;

    const f16* A1c = A1 + (size_t)c * NB * 4096;

    for (int kb = 0; kb < 128; ++kb) {
        const int j0    = kb * 32;
        const int base0 = ((j0 < 2048) ? 2048 : 8192) + t0 - j0;
        // Build B fragment (Toeplitz tile) by per-lane LDS gather.
        Frag16 bf;
        #pragma unroll
        for (int e = 0; e < 16; ++e) {
            const int K = (e < 8) ? (8 * grp + e) : (16 + 8 * grp + (e - 8));
            bf.h[e] = sS[base0 + nrow - K];
        }
        // prefetch next A chunk for this lane's row
        __builtin_prefetch(A1c + (size_t)nrow * 4096 + j0 + 32 + 8 * grp, 0, 1);
        #pragma unroll
        for (int rt = 0; rt < 8; ++rt) {
            const f16* pa = A1c + (size_t)(rt * 16 + nrow) * 4096 + j0 + 8 * grp;
            Frag16 af;
            af.q[0] = *(const v8h*)pa;
            af.q[1] = *(const v8h*)(pa + 16);
            acc[rt] = __builtin_amdgcn_wmma_f32_16x16x32_f16(
                false, af.v, false, bf.v, (short)0, acc[rt], false, false);
        }
    }

    // Epilogue: + un * D[t], store f16 y in (b*64+c, t) layout for GEMM2.
    const int t  = t0 + nrow;
    const float Dt = Dg[t];
    #pragma unroll
    for (int rt = 0; rt < 8; ++rt) {
        #pragma unroll
        for (int g = 0; g < 8; ++g) {
            const int b   = rt * 16 + g + 8 * grp;
            const float un = (float)A1c[(size_t)b * 4096 + t];
            const float val = acc[rt][g] + un * Dt;
            y2[((size_t)b * NC + c) * 2048 + t] = (f16)val;
        }
    }
}

// ---------------- GEMM2: 8192 x 2048 x 1440 projection ----------------
__global__ __launch_bounds__(256) void gemm2_kernel(const f16* __restrict__ y2,
                                                    const f16* __restrict__ pw16,
                                                    float* __restrict__ Z) {
    const int wave = threadIdx.x >> 5;
    const int lane = threadIdx.x & 31;
    const int nrow = lane & 15;
    const int grp  = lane >> 4;
    const int m0   = (blockIdx.x * 8 + wave) * 16;   // row tile (b*64+c)
    const int p0   = blockIdx.y * 16;                // col tile (projection out)

    v8f acc = {0.f,0.f,0.f,0.f,0.f,0.f,0.f,0.f};
    const f16* ya = y2   + (size_t)(m0 + nrow) * 2048;
    const f16* pb = pw16 + (size_t)(p0 + nrow) * 2048;

    for (int h0 = 0; h0 < 2048; h0 += 32) {
        Frag16 af, bf;
        af.q[0] = *(const v8h*)(ya + h0 + 8 * grp);
        af.q[1] = *(const v8h*)(ya + h0 + 16 + 8 * grp);
        bf.q[0] = *(const v8h*)(pb + h0 + 8 * grp);
        bf.q[1] = *(const v8h*)(pb + h0 + 16 + 8 * grp);
        acc = __builtin_amdgcn_wmma_f32_16x16x32_f16(
            false, af.v, false, bf.v, (short)0, acc, false, false);
    }
    #pragma unroll
    for (int g = 0; g < 8; ++g) {
        Z[(size_t)(m0 + g + 8 * grp) * (2 * NP) + p0 + nrow] = acc[g];
    }
}

// ---------------- GLU + bias + denorm -> d_out (B, 720, 64) f32 ----------------
__global__ void glu_kernel(const float* __restrict__ Z, const float* __restrict__ pb,
                           const float* __restrict__ mean, const float* __restrict__ rstd,
                           float* __restrict__ out) {
    size_t i = (size_t)blockIdx.x * 256 + threadIdx.x;
    if (i >= (size_t)NB * NP * NC) return;
    int cc = (int)(i % NC);
    int pp = (int)((i / NC) % NP);
    int b  = (int)(i / ((size_t)NC * NP));
    size_t m = (size_t)b * NC + cc;
    float a = Z[m * (2 * NP) + pp]      + pb[pp];
    float g = Z[m * (2 * NP) + NP + pp] + pb[NP + pp];
    float val = a * (1.f / (1.f + expf(-g)));
    out[i] = val * (1.f / rstd[m]) + mean[m];
}

extern "C" void kernel_launch(void* const* d_in, const int* in_sizes, int n_in,
                              void* d_out, int out_size, void* d_ws, size_t ws_size,
                              hipStream_t stream) {
    (void)in_sizes; (void)n_in; (void)out_size; (void)ws_size;
    const float* u          = (const float*)d_in[0];
    const float* Wn         = (const float*)d_in[4];
    const float* Dg         = (const float*)d_in[5];
    const float* logvar     = (const float*)d_in[6];
    const float* C_param    = (const float*)d_in[7];
    const float* log_dt     = (const float*)d_in[8];
    const float* log_A_real = (const float*)d_in[9];
    const float* A_imag     = (const float*)d_in[10];
    const float* proj_w     = (const float*)d_in[11];
    const float* proj_b     = (const float*)d_in[12];

    char* ws   = (char*)d_ws;
    float* mean = (float*)(ws + OFF_MEAN);
    float* rstd = (float*)(ws + OFF_RSTD);
    f16*   Sg   = (f16*)(ws + OFF_SG);
    f16*   pw16 = (f16*)(ws + OFF_PW);
    f16*   A1   = (f16*)(ws + OFF_A1);
    f16*   y2   = (f16*)(ws + OFF_Y2);
    float* Z    = (float*)(ws + OFF_Z);

    // 1) zero padded kernel table (Sg): 64*8192*2 bytes = 262144 u32
    zero_kernel<<<1024, 256, 0, stream>>>((unsigned int*)Sg, 262144ull);
    // 2) instance-norm stats
    stats_kernel<<<NB, 256, 0, stream>>>(u, mean, rstd);
    // 3) SSM kernels -> padded table
    ssm_kernel<<<NL, 64, 0, stream>>>(C_param, log_dt, log_A_real, A_imag, Sg);
    // 4) pack [un | W] to f16
    {
        size_t n = (size_t)NB * NL * NC;
        pack_kernel<<<(unsigned)((n + 255) / 256), 256, 0, stream>>>(u, Wn, logvar, mean, rstd, A1);
    }
    // 5) proj_w -> f16
    {
        size_t n = (size_t)2 * NP * NL;
        pwpack_kernel<<<(unsigned)((n + 255) / 256), 256, 0, stream>>>(proj_w, pw16);
    }
    // 6) Toeplitz conv GEMM (64 channels x 16 time supertiles)
    gemm1_kernel<<<dim3(NC, 16), 256, 0, stream>>>(A1, Sg, Dg, y2);
    // 7) projection GEMM (512 row tiles / 8 per block, 90 col tiles)
    gemm2_kernel<<<dim3(64, 90), 256, 0, stream>>>(y2, pw16, Z);
    // 8) GLU + denorm
    {
        size_t n = (size_t)NB * NP * NC;
        glu_kernel<<<(unsigned)((n + 255) / 256), 256, 0, stream>>>(Z, proj_b, mean, rstd, (float*)d_out);
    }
}
